// BinaryLinear_16707422781541
// MI455X (gfx1250) — matrix operand order repaired and checked by reading, not compiled, not run on hardware
//
#include <hip/hip_runtime.h>
#include <hip/hip_bf16.h>

// ---------------------------------------------------------------------------
// BinaryLinear: out = (x @ (sign(w)*alpha)^T), alpha = mean(|w|, axis=in)
// f16 WMMA GEMM with exact +/-1 weights, f32 accumulate, alpha epilogue scale.
// Double-buffered LDS, async global->LDS for pre-packed B fragments.
// Wave tile 64x32 (4 A-frags x 2 B-frags -> 8 WMMA, 2 wait groups/K-step).
// ---------------------------------------------------------------------------

typedef __attribute__((ext_vector_type(16))) _Float16 v16h;
typedef __attribute__((ext_vector_type(8)))  _Float16 v8h;
typedef __attribute__((ext_vector_type(8)))  float    v8f;
typedef __attribute__((ext_vector_type(4)))  float    v4f;
typedef __attribute__((ext_vector_type(4)))  int      v4i;

#define TOKENS 65536
#define IN_F   1024
#define OUT_F  1024

#define KC_STEPS    (IN_F / 32)    // 32 K-chunks of 32
#define LDSX_STRIDE 40             // halves per x row in LDS (80B, read-conflict-free)

// ---------------------------------------------------------------------------
// Kernel 1: alpha[o] = mean(|w[o,:]|). One wave per output row.
// ---------------------------------------------------------------------------
__global__ __launch_bounds__(256) void alpha_kernel(const float* __restrict__ w,
                                                    float* __restrict__ alpha) {
    const int lane = threadIdx.x & 31;
    const int wave = threadIdx.x >> 5;
    const int o = blockIdx.x * 8 + wave;
    float s = 0.0f;
#pragma unroll
    for (int i = 0; i < IN_F / 32; ++i)
        s += fabsf(w[(size_t)o * IN_F + lane + i * 32]);
#pragma unroll
    for (int off = 16; off > 0; off >>= 1)
        s += __shfl_xor(s, off, 32);
    if (lane == 0) alpha[o] = s * (1.0f / (float)IN_F);
}

// ---------------------------------------------------------------------------
// Kernel 2: pack sign(w) as +/-1 f16 in WMMA B-fragment layout.
// B fragment (32x16 KxN), contiguous k order (same as A): lane<16 -> N=lane, K elems 0..15 ; lane>=16 ->
// N=lane-16, K elems 16..31.
// Flat order (bn, kc, s, lane, e): f = ((((bn*32 + kc)*8 + s)*32 + lane)*16 + e
// so each block's per-K 8KB chunk is contiguous.
// ---------------------------------------------------------------------------
__global__ __launch_bounds__(256) void pack_kernel(const float* __restrict__ w,
                                                   _Float16* __restrict__ bsign) {
    const int f    = blockIdx.x * 256 + threadIdx.x;   // 1M elements
    const int e    = f & 15;
    const int lane = (f >> 4) & 31;
    const int s    = (f >> 9) & 7;
    const int kc   = (f >> 12) & 31;
    const int bn   = f >> 17;                          // 0..7
    const int o    = (bn * 8 + s) * 16 + (lane & 15);
    const int k    = kc * 32 + e + ((lane & 16) ? 16 : 0);
    const float wv = w[(size_t)o * IN_F + k];
    _Float16 sgn = (wv > 0.0f) ? (_Float16)1.0f
                 : ((wv < 0.0f) ? (_Float16)-1.0f : (_Float16)0.0f);
    bsign[f] = sgn;
}

// ---------------------------------------------------------------------------
// async copy helpers (guarded; fall back to synchronous b128 copies)
// ---------------------------------------------------------------------------
__device__ __forceinline__ void async_b128(const void* gsrc, void* ldst) {
#if __has_builtin(__builtin_amdgcn_global_load_async_to_lds_b128)
    __builtin_amdgcn_global_load_async_to_lds_b128(
        (__attribute__((address_space(1))) v4i*)gsrc,
        (__attribute__((address_space(3))) v4i*)ldst,
        0, 0);
#else
    *(v4f*)ldst = *(const v4f*)gsrc;
#endif
}

__device__ __forceinline__ void wait_async0() {
#if __has_builtin(__builtin_amdgcn_global_load_async_to_lds_b128)
#if __has_builtin(__builtin_amdgcn_s_wait_asynccnt)
    __builtin_amdgcn_s_wait_asynccnt(0);
#else
    asm volatile("s_wait_asynccnt 0x0" ::: "memory");
#endif
#endif
}

// ---------------------------------------------------------------------------
// Kernel 3: GEMM. 256 threads (8 waves), block tile = 128 rows x 128 cols,
// double-buffered K pipeline. Wave grid 2x4: wave owns 64 rows x 32 cols.
// ---------------------------------------------------------------------------
__global__ __launch_bounds__(256, 2) void gemm_kernel(const float* __restrict__ x,
                                                      const _Float16* __restrict__ bsign,
                                                      const float* __restrict__ alpha,
                                                      float* __restrict__ out) {
    __shared__ __align__(16) _Float16 lds_x[2][128 * LDSX_STRIDE];  // 2 x 10 KB
    __shared__ __align__(16) _Float16 lds_b[2][8 * 32 * 16];        // 2 x  8 KB

    const int t    = threadIdx.x;
    const int lane = t & 31;
    const int wave = t >> 5;
    const int bm   = blockIdx.x;           // 0..511  (M tiles of 128)
    const int bn   = blockIdx.y;           // 0..7    (N tiles of 128)
    const size_t rowBase = (size_t)bm * 128;

    v8f acc[8] = {};                       // acc[s*4 + mi]

    // staging geometry: thread -> 8 contiguous floats, twice (128x32 chunk)
    const int xr0 = t >> 2;                // rows 0..63  (+64 on 2nd pass)
    const int xc  = (t & 3) * 8;           // col 0,8,16,24

    auto stage = [&](int p, int kc) {
        // ---- x chunk: 2 x (2 global b128 -> packed cvt -> 1 ds b128) ----
#pragma unroll
        for (int it = 0; it < 2; ++it) {
            const int r = xr0 + it * 64;
            const v4f* gp = (const v4f*)&x[(rowBase + r) * IN_F + (size_t)kc * 32 + xc];
            union { v8f v; v4f q[2]; } f;
            f.q[0] = gp[0];
            f.q[1] = gp[1];
            v8h h = __builtin_convertvector(f.v, v8h);
            *(v8h*)&lds_x[p][r * LDSX_STRIDE + xc] = h;
        }
        // ---- B fragments: contiguous 8 KB chunk, async global->LDS ----
        const char* src = (const char*)bsign + ((size_t)(bn * 32 + kc) * 8192) + t * 32;
        char* dst = (char*)&lds_b[p][0] + t * 32;
        async_b128(src, dst);
        async_b128(src + 16, dst + 16);
    };

    // Wave tile: rows [mrow0, mrow0+64), cols [(wn*2)*16, +32)
    const int wm    = wave & 1;
    const int wn    = wave >> 1;           // 0..3
    const int mrow0 = wm * 64;
    const int sg    = wn * 2;              // B subtile group start (0,2,4,6)

    // A-fragment addressing (16x32, contiguous k order, same as packed B):
    //   lane<16: row=lane,    K 0..15
    //   lane>=16: row=lane-16, K 16..31
    const int klo   = (lane & 16) ? 16 : 0;
    const int aoff0 = (mrow0 + (lane & 15)) * LDSX_STRIDE + klo;

    stage(0, 0);
    int p = 0;
    for (int kc = 0; kc < KC_STEPS; ++kc) {
        wait_async0();        // this wave's async B for buffer p has landed
        __syncthreads();      // all waves' staging of buffer p visible

        // ---- load ALL fragments for this iteration up front (12 ds_b128);
        //      DS is in-order, so waits can be partial and WMMAs pipeline ----
        union F16x16 { v16h v; v8h h[2]; };
        F16x16 a[4], bf[2];
#pragma unroll
        for (int mi = 0; mi < 4; ++mi) {
            const int ao = aoff0 + mi * 16 * LDSX_STRIDE;
            a[mi].h[0] = *(const v8h*)&lds_x[p][ao];
            a[mi].h[1] = *(const v8h*)&lds_x[p][ao + 8];
        }
#pragma unroll
        for (int s = 0; s < 2; ++s) {
            const v8h* bp = (const v8h*)&lds_b[p][((sg + s) * 32 + lane) * 16];
            bf[s].h[0] = bp[0];
            bf[s].h[1] = bp[1];
        }

        // stage next chunk while fragments are in flight / WMMAs run
        if (kc + 1 < KC_STEPS) stage(p ^ 1, kc + 1);

        // 2 wait groups x 4 back-to-back WMMAs
#pragma unroll
        for (int s = 0; s < 2; ++s) {
#pragma unroll
            for (int mi = 0; mi < 4; ++mi) {
                acc[s * 4 + mi] = __builtin_amdgcn_wmma_f32_16x16x32_f16(
                    false, a[mi].v, false, bf[s].v, (short)0, acc[s * 4 + mi],
                    false, false);
            }
        }
        p ^= 1;
    }

    // ---- epilogue: scale by alpha[o], nontemporal f32 store ----
    // C layout: lane<16 -> N=lane, M=v ; lane>=16 -> N=lane-16, M=v+8
#pragma unroll
    for (int s = 0; s < 2; ++s) {
        const int o = bn * 128 + (sg + s) * 16 + (lane & 15);
        const float al = alpha[o];
#pragma unroll
        for (int mi = 0; mi < 4; ++mi) {
            const size_t r0 = rowBase + mrow0 + mi * 16 + ((lane & 16) ? 8 : 0);
#pragma unroll
            for (int v = 0; v < 8; ++v) {
                __builtin_nontemporal_store(acc[s * 4 + mi][v] * al,
                                            &out[(r0 + v) * OUT_F + o]);
            }
        }
    }
}

// ---------------------------------------------------------------------------
extern "C" void kernel_launch(void* const* d_in, const int* in_sizes, int n_in,
                              void* d_out, int out_size, void* d_ws, size_t ws_size,
                              hipStream_t stream) {
    const float* x = (const float*)d_in[0];   // [65536, 1024] f32
    const float* w = (const float*)d_in[1];   // [1024, 1024] f32
    float* out = (float*)d_out;               // [65536, 1024] f32

    // workspace: alpha (4 KB) | packed sign weights (2 MB, f16 fragment layout)
    float*    ws_alpha = (float*)d_ws;
    _Float16* ws_bsign = (_Float16*)((char*)d_ws + 4096);

    alpha_kernel<<<128, 256, 0, stream>>>(w, ws_alpha);
    pack_kernel<<<(OUT_F * IN_F) / 256, 256, 0, stream>>>(w, ws_bsign);
    gemm_kernel<<<dim3(TOKENS / 128, OUT_F / 128, 1), 256, 0, stream>>>(
        x, ws_bsign, ws_alpha, out);
}
